// PositionAwareSparseAttention_71717363908873
// MI455X (gfx1250) — matrix operand order repaired and checked by reading, not compiled, not run on hardware
//
#include <hip/hip_runtime.h>
#include <hip/hip_bf16.h>
#include <math.h>

// ---------------------------------------------------------------------------
// PositionAwareSparseAttention for MI455X (gfx1250, wave32, WMMA 16x16x32 f16)
//
// Pipeline:
//   1) convert x -> f16, transpose+convert Wq/Wk/Wv/Wo -> f16 [out][in]
//   2) fused QKV GEMM via v_wmma_f32_16x16x32_f16 (Q,K stored [B,N,128] f16,
//      V stored transposed [B,H,D,N] f16 so P*V B-fragments load contiguously)
//   3) flash attention: per-wave 16-row q tile, 32-key steps, online softmax,
//      T5 relative-position bias, probabilities staged through LDS to convert
//      C-layout -> A-layout, accumulating WMMAs for P*V
//   4) output projection GEMM (WMMA) + bias -> f32 d_out
//
// Workspace: ~21.1 MB.
// ---------------------------------------------------------------------------

typedef __attribute__((ext_vector_type(16))) _Float16 v16h;
typedef __attribute__((ext_vector_type(8)))  float    v8f;

#define LATENT   128
#define HEADS    4
#define HEAD_DIM 32
#define NBUCK    32
#define B_       8
#define N_       2048
#define M_TOK    (B_ * N_)

// A-matrix (16-bit, 16x32) element -> K index. lane: row = lane&15, half = lane>>4.
__device__ __forceinline__ int a_kidx(int i, int half) {
  return i + (half << 4);
}

__device__ __forceinline__ v8f wmma_f16(v16h a, v16h b, v8f c) {
  return __builtin_amdgcn_wmma_f32_16x16x32_f16(false, a, false, b,
                                                (short)0, c, false, false);
}

// ---------------------------------------------------------------------------
// Prep kernels
// ---------------------------------------------------------------------------
__global__ __launch_bounds__(256) void k_f32_to_f16(const float* __restrict__ src,
                                                    _Float16* __restrict__ dst,
                                                    int n) {
  int i = blockIdx.x * blockDim.x + threadIdx.x;
  int stride = gridDim.x * blockDim.x;
  for (; i < n; i += stride) dst[i] = (_Float16)src[i];
}

// W [128 in][128 out] f32 -> Wt [128 out][128 in] f16
__global__ __launch_bounds__(256) void k_transpose_w(const float* __restrict__ W,
                                                     _Float16* __restrict__ Wt) {
  int i = blockIdx.x * blockDim.x + threadIdx.x;  // 0..16383
  int k = i >> 7, n = i & 127;
  Wt[n * 128 + k] = (_Float16)W[k * 128 + n];
}

// ---------------------------------------------------------------------------
// Fused QKV projection: [16384,128] x [128,128] x3 via WMMA.
// Block = 256 threads = 8 waves; wave w owns output n-tile w (16 cols) for all
// three weight matrices. Grid.x = M/16.
// ---------------------------------------------------------------------------
__global__ __launch_bounds__(256) void k_qkv(
    const _Float16* __restrict__ xh,
    const _Float16* __restrict__ Wqt, const _Float16* __restrict__ Wkt,
    const _Float16* __restrict__ Wvt,
    const float* __restrict__ bq, const float* __restrict__ bk,
    const float* __restrict__ bv,
    _Float16* __restrict__ Qh, _Float16* __restrict__ Kh,
    _Float16* __restrict__ Vt) {
  const int lane = threadIdx.x & 31;
  const int wave = threadIdx.x >> 5;
  const int m0   = blockIdx.x * 16;
  const int row  = lane & 15;   // A row / B col / C col
  const int half = lane >> 4;

  // A fragments for the 16x128 x-tile (4 k-steps of 32)
  v16h a[4];
  const _Float16* xrow = xh + (m0 + row) * 128;
#pragma unroll
  for (int ks = 0; ks < 4; ++ks)
#pragma unroll
    for (int i = 0; i < 16; ++i)
      a[ks][i] = xrow[ks * 32 + a_kidx(i, half)];

  const int n0 = wave * 16;
#pragma unroll
  for (int w = 0; w < 3; ++w) {
    const _Float16* Wt  = (w == 0) ? Wqt : (w == 1) ? Wkt : Wvt;
    const float* biasv  = (w == 0) ? bq  : (w == 1) ? bk  : bv;
    v8f c = {0.f, 0.f, 0.f, 0.f, 0.f, 0.f, 0.f, 0.f};
#pragma unroll
    for (int ks = 0; ks < 4; ++ks) {
      // B[k][n] = W[k][n] = Wt[n][k]; lane col = row, k = half*16 + i (contig)
      const _Float16* wrow = Wt + (n0 + row) * 128 + ks * 32 + half * 16;
      v16h bf;
#pragma unroll
      for (int i = 0; i < 16; ++i) bf[i] = wrow[i];
      c = wmma_f16(a[ks], bf, c);
    }
    const int col = n0 + row;
    const float bb = biasv[col];
#pragma unroll
    for (int r = 0; r < 8; ++r) {
      const int tok = m0 + r + 8 * half;
      const float val = c[r] + bb;
      if (w == 0) {
        Qh[tok * 128 + col] = (_Float16)val;
      } else if (w == 1) {
        Kh[tok * 128 + col] = (_Float16)val;
      } else {
        const int bi = tok >> 11, nn = tok & (N_ - 1);
        const int h = col >> 5, d = col & 31;
        Vt[((bi * HEADS + h) * HEAD_DIM + d) * N_ + nn] = (_Float16)val;
      }
    }
  }
}

// T5 relative position bucket -> bias (pos_w column for head h staged in LDS)
__device__ __forceinline__ float rel_bias(int qp, int kp,
                                          const float* __restrict__ s_posw) {
  const int rel = kp - qp;
  int bucket = (rel > 0) ? 16 : 0;
  const int arel = (rel < 0) ? -rel : rel;
  if (arel < 8) {
    bucket += arel;
  } else {
    // 8 + log(arel/8)/log(12500) * 8 ; 8/ln(12500) = 0.848044451
    int lg = 8 + (int)(__logf((float)arel * 0.125f) * 0.848044451f);
    bucket += (lg < 15) ? lg : 15;
  }
  return s_posw[bucket];
}

// ---------------------------------------------------------------------------
// Flash attention. Block = 8 waves; each wave owns one 16-row q tile of one
// (b,h). Grid = (N/128, HEADS, B). Key loop steps 32 (two 16-key WMMAs).
// ---------------------------------------------------------------------------
__global__ __launch_bounds__(256) void k_flash(
    const _Float16* __restrict__ Qh, const _Float16* __restrict__ Kh,
    const _Float16* __restrict__ Vt,
    const int* __restrict__ positions, const unsigned char* __restrict__ mask,
    const float* __restrict__ pos_w, _Float16* __restrict__ att) {
  __shared__ float    s_posw[NBUCK + 1];
  __shared__ _Float16 sP[8][16][40];  // per-wave P tile staging (padded rows)

  const int b    = blockIdx.z;
  const int h    = blockIdx.y;
  const int lane = threadIdx.x & 31;
  const int wave = threadIdx.x >> 5;
  const int q0   = (blockIdx.x * 8 + wave) * 16;
  const int col  = lane & 15;
  const int half = lane >> 4;

  if (threadIdx.x < NBUCK + 1)
    s_posw[threadIdx.x] = pos_w[threadIdx.x * HEADS + h];
  __syncthreads();

  // Q A-fragment: 16 q rows x 32 head-dim
  v16h aq;
  const _Float16* qrow = Qh + (size_t)(b * N_ + q0 + col) * 128 + h * HEAD_DIM;
#pragma unroll
  for (int i = 0; i < 16; ++i) aq[i] = qrow[a_kidx(i, half)];

  int qp[8];
#pragma unroll
  for (int r = 0; r < 8; ++r)
    qp[r] = positions[b * N_ + q0 + r + 8 * half];

  float m[8], l[8];
#pragma unroll
  for (int r = 0; r < 8; ++r) { m[r] = -__builtin_inff(); l[r] = 0.f; }
  v8f o0 = {0.f, 0.f, 0.f, 0.f, 0.f, 0.f, 0.f, 0.f};
  v8f o1 = {0.f, 0.f, 0.f, 0.f, 0.f, 0.f, 0.f, 0.f};
  const v8f zero = {0.f, 0.f, 0.f, 0.f, 0.f, 0.f, 0.f, 0.f};
  const float scale = 0.17677669529663687f;  // 1/sqrt(32)

  for (int j0 = 0; j0 < N_; j0 += 32) {
    // K^T B-fragments: B[d][key] = K[j0+key][d]; contiguous in d per lane
    const _Float16* kr0 =
        Kh + (size_t)(b * N_ + j0 + col) * 128 + h * HEAD_DIM + half * 16;
    const _Float16* kr1 = kr0 + 16 * 128;
    v16h bk0, bk1;
#pragma unroll
    for (int i = 0; i < 16; ++i) { bk0[i] = kr0[i]; bk1[i] = kr1[i]; }
    v8f s0 = wmma_f16(aq, bk0, zero);
    v8f s1 = wmma_f16(aq, bk1, zero);

    const int kp0 = positions[b * N_ + j0 + col];
    const int kp1 = positions[b * N_ + j0 + 16 + col];
    const bool km0 = mask[b * N_ + j0 + col] != 0;
    const bool km1 = mask[b * N_ + j0 + 16 + col] != 0;

#pragma unroll
    for (int r = 0; r < 8; ++r) {
      float v0 = km0 ? s0[r] * scale + rel_bias(qp[r], kp0, s_posw)
                     : -__builtin_inff();
      float v1 = km1 ? s1[r] * scale + rel_bias(qp[r], kp1, s_posw)
                     : -__builtin_inff();
      // row max across the 16 lanes holding this row
      float t = fmaxf(v0, v1);
      t = fmaxf(t, __shfl_xor(t, 1, 32));
      t = fmaxf(t, __shfl_xor(t, 2, 32));
      t = fmaxf(t, __shfl_xor(t, 4, 32));
      t = fmaxf(t, __shfl_xor(t, 8, 32));
      float nm = fmaxf(m[r], t);
      nm = fmaxf(nm, -3.0e38f);  // all-masked row: exp(x-nm)=0, l stays 0
      const float alpha = __expf(m[r] - nm);
      const float e0 = __expf(v0 - nm);
      const float e1 = __expf(v1 - nm);
      float rs = e0 + e1;
      rs += __shfl_xor(rs, 1, 32);
      rs += __shfl_xor(rs, 2, 32);
      rs += __shfl_xor(rs, 4, 32);
      rs += __shfl_xor(rs, 8, 32);
      l[r] = l[r] * alpha + rs;
      m[r] = nm;
      o0[r] *= alpha;
      o1[r] *= alpha;
      // stash probabilities (C layout -> LDS row-major)
      sP[wave][r + 8 * half][col]      = (_Float16)e0;
      sP[wave][r + 8 * half][16 + col] = (_Float16)e1;
    }
    asm volatile("s_wait_dscnt 0" ::: "memory");

    // reload P as A-fragment (16 q rows x 32 keys)
    v16h ap;
#pragma unroll
    for (int i = 0; i < 16; ++i) ap[i] = sP[wave][col][a_kidx(i, half)];

    // V B-fragments: B[k][d] = Vt[b,h,d][j0+k]; contiguous in k per lane
    const _Float16* vr0 =
        Vt + (size_t)((b * HEADS + h) * HEAD_DIM + col) * N_ + j0 + half * 16;
    const _Float16* vr1 = vr0 + 16 * N_;
    v16h bv0, bv1;
#pragma unroll
    for (int i = 0; i < 16; ++i) { bv0[i] = vr0[i]; bv1[i] = vr1[i]; }
    o0 = wmma_f16(ap, bv0, o0);
    o1 = wmma_f16(ap, bv1, o1);
  }

#pragma unroll
  for (int r = 0; r < 8; ++r) {
    const float inv = (l[r] > 1e-30f) ? 1.0f / l[r] : 0.0f;  // nan_to_num path
    const size_t tok = (size_t)(b * N_ + q0 + r + 8 * half);
    att[tok * 128 + h * HEAD_DIM + col]      = (_Float16)(o0[r] * inv);
    att[tok * 128 + h * HEAD_DIM + 16 + col] = (_Float16)(o1[r] * inv);
  }
}

// ---------------------------------------------------------------------------
// Output projection: att [16384,128] f16 x Wo [128,128] + bo -> f32 out
// ---------------------------------------------------------------------------
__global__ __launch_bounds__(256) void k_outproj(
    const _Float16* __restrict__ atth, const _Float16* __restrict__ Wot,
    const float* __restrict__ bo, float* __restrict__ out) {
  const int lane = threadIdx.x & 31;
  const int wave = threadIdx.x >> 5;
  const int m0   = blockIdx.x * 16;
  const int row  = lane & 15;
  const int half = lane >> 4;

  v16h a[4];
  const _Float16* arow = atth + (size_t)(m0 + row) * 128;
#pragma unroll
  for (int ks = 0; ks < 4; ++ks)
#pragma unroll
    for (int i = 0; i < 16; ++i)
      a[ks][i] = arow[ks * 32 + a_kidx(i, half)];

  const int n0 = wave * 16;
  v8f c = {0.f, 0.f, 0.f, 0.f, 0.f, 0.f, 0.f, 0.f};
#pragma unroll
  for (int ks = 0; ks < 4; ++ks) {
    const _Float16* wrow = Wot + (n0 + row) * 128 + ks * 32 + half * 16;
    v16h bf;
#pragma unroll
    for (int i = 0; i < 16; ++i) bf[i] = wrow[i];
    c = wmma_f16(a[ks], bf, c);
  }
  const int col = n0 + row;
  const float bb = bo[col];
#pragma unroll
  for (int r = 0; r < 8; ++r)
    out[(size_t)(m0 + r + 8 * half) * 128 + col] = c[r] + bb;
}

// ---------------------------------------------------------------------------
extern "C" void kernel_launch(void* const* d_in, const int* in_sizes, int n_in,
                              void* d_out, int out_size, void* d_ws,
                              size_t ws_size, hipStream_t stream) {
  const float*         x         = (const float*)d_in[0];
  const int*           positions = (const int*)d_in[1];
  const unsigned char* mask      = (const unsigned char*)d_in[2];
  const float*         Wq        = (const float*)d_in[3];
  const float*         bq        = (const float*)d_in[4];
  const float*         Wk        = (const float*)d_in[5];
  const float*         bk        = (const float*)d_in[6];
  const float*         Wv        = (const float*)d_in[7];
  const float*         bv        = (const float*)d_in[8];
  const float*         pos_w     = (const float*)d_in[9];
  const float*         Wo        = (const float*)d_in[10];
  const float*         bo        = (const float*)d_in[11];
  float*               out       = (float*)d_out;

  char* ws = (char*)d_ws;
  size_t off = 0;
  auto alloc = [&](size_t bytes) -> void* {
    void* p = ws + off;
    off = (off + bytes + 255) & ~(size_t)255;
    return p;
  };
  _Float16* xh   = (_Float16*)alloc((size_t)M_TOK * 128 * 2);  // 4 MB
  _Float16* Qh   = (_Float16*)alloc((size_t)M_TOK * 128 * 2);  // 4 MB
  _Float16* Kh   = (_Float16*)alloc((size_t)M_TOK * 128 * 2);  // 4 MB
  _Float16* Vt   = (_Float16*)alloc((size_t)M_TOK * 128 * 2);  // 4 MB [B,H,D,N]
  _Float16* atth = (_Float16*)alloc((size_t)M_TOK * 128 * 2);  // 4 MB
  _Float16* Wqt  = (_Float16*)alloc(128 * 128 * 2);
  _Float16* Wkt  = (_Float16*)alloc(128 * 128 * 2);
  _Float16* Wvt  = (_Float16*)alloc(128 * 128 * 2);
  _Float16* Wot  = (_Float16*)alloc(128 * 128 * 2);

  k_f32_to_f16<<<4096, 256, 0, stream>>>(x, xh, M_TOK * 128);
  k_transpose_w<<<64, 256, 0, stream>>>(Wq, Wqt);
  k_transpose_w<<<64, 256, 0, stream>>>(Wk, Wkt);
  k_transpose_w<<<64, 256, 0, stream>>>(Wv, Wvt);
  k_transpose_w<<<64, 256, 0, stream>>>(Wo, Wot);

  k_qkv<<<M_TOK / 16, 256, 0, stream>>>(xh, Wqt, Wkt, Wvt, bq, bk, bv,
                                        Qh, Kh, Vt);

  k_flash<<<dim3(N_ / 128, HEADS, B_), 256, 0, stream>>>(
      Qh, Kh, Vt, positions, mask, pos_w, atth);

  k_outproj<<<M_TOK / 16, 256, 0, stream>>>(atth, Wot, bo, out);
}
